// AttentionDecoder_47510928229113
// MI455X (gfx1250) — matrix operand order repaired and checked by reading, not compiled, not run on hardware
//
#include <hip/hip_runtime.h>
#include <hip/hip_bf16.h>
#include <math.h>

// Problem constants
#define Bn   64
#define Sn   128
#define En   1024
#define Hn   1024
#define An   512
#define EMBn 512
#define Vn   32000

typedef __attribute__((ext_vector_type(16))) __bf16 v16bf;
typedef __attribute__((ext_vector_type(8)))  float  v8f;

// ---------------------------------------------------------------------------
// Branchless fast transcendentals (v_exp_f32 / v_rcp_f32, no EXEC divergence)
// ---------------------------------------------------------------------------
__device__ __forceinline__ float rcp_fast(float x) { return __builtin_amdgcn_rcpf(x); }

__device__ __forceinline__ float tanh_fast(float x) {
    const float xc = fminf(fmaxf(x, -15.0f), 15.0f);   // avoid inf/inf
    const float e  = __expf(2.0f * xc);
    return (e - 1.0f) * rcp_fast(e + 1.0f);
}

__device__ __forceinline__ float sigmoid_fast(float x) {
    return rcp_fast(1.0f + __expf(-x));                // exp overflow -> 0/1, correct
}

// ---------------------------------------------------------------------------
// bf16 pack helpers following CDNA5 WMMA VGPR layouts (wave32).
// A 16x32 (MxK): lane = half*16 + m; elems 0..7 -> K = 8*half + i,
//                elems 8..15 -> K = 16 + 8*half + (i-8)
// B 32x16 (KxN): lane = half*16 + n; same K mapping as A
// D 16x16:       elem j -> row = j + 8*half, col = lane & 15
// ---------------------------------------------------------------------------
__device__ __forceinline__ void cvt4(const float4 f, v16bf& d, int o) {
    d[o + 0] = (__bf16)f.x;
    d[o + 1] = (__bf16)f.y;
    d[o + 2] = (__bf16)f.z;
    d[o + 3] = (__bf16)f.w;
}

__device__ __forceinline__ void load_a16(const float* __restrict__ row, int k, int half, v16bf& a) {
    const float* p0 = row + k + 8 * half;
    const float* p1 = row + k + 16 + 8 * half;
    cvt4(*(const float4*)(p0),     a, 0);
    cvt4(*(const float4*)(p0 + 4), a, 4);
    cvt4(*(const float4*)(p1),     a, 8);
    cvt4(*(const float4*)(p1 + 4), a, 12);
}

__device__ __forceinline__ void load_b16(const float* __restrict__ row, int k, int half, v16bf& b) {
    const float* p = row + k + 8 * half;
    cvt4(*(const float4*)(p),      b, 0);
    cvt4(*(const float4*)(p + 4),  b, 4);
    cvt4(*(const float4*)(p + 16), b, 8);
    cvt4(*(const float4*)(p + 20), b, 12);
}

// ---------------------------------------------------------------------------
// fp32-in / fp32-out GEMM via bf16 WMMA, f32 accumulate.
// C[M,N] (+)= A[M,K] @ W[N,K]^T + bias[N],  M = MT*16 exactly.
// One wave owns one 16-column N-tile and ALL MT row-tiles: each B fragment is
// loaded from HBM once and reused for MT WMMAs, so big weight matrices are
// streamed exactly once (the bandwidth floor on this memory-bound problem).
// blockDim = 256 (8 waves). Requires N mult of 16, K mult of 32.
// ---------------------------------------------------------------------------
template <int MT>
__global__ void gemm_bias_wmma(const float* __restrict__ A, int lda,
                               const float* __restrict__ W, int ldw,
                               const float* __restrict__ bias,
                               float* __restrict__ C, int ldc,
                               int N, int K, int accumulate)
{
    const int wave = (blockIdx.x * blockDim.x + threadIdx.x) >> 5;
    const int lane = threadIdx.x & 31;
    const int half = lane >> 4;
    const int l    = lane & 15;

    const int n0 = wave << 4;
    if (n0 >= N) return;

    const float* wrow = W + (size_t)(n0 + l) * (size_t)ldw;
    const float* arow[MT];
#pragma unroll
    for (int t = 0; t < MT; ++t)
        arow[t] = A + (size_t)(t * 16 + l) * (size_t)lda;

    v8f acc[MT];
#pragma unroll
    for (int t = 0; t < MT; ++t) acc[t] = (v8f){};

#pragma unroll 1
    for (int k = 0; k < K; k += 32) {
        __builtin_prefetch(wrow + k + 64, 0, 1);   // global_prefetch next weight block
        v16bf b;
        load_b16(wrow, k, half, b);
#pragma unroll
        for (int t = 0; t < MT; ++t) {
            v16bf a;
            load_a16(arow[t], k, half, a);
            acc[t] = __builtin_amdgcn_wmma_f32_16x16x32_bf16(
                         false, a, false, b, (short)0, acc[t], false, false);
        }
    }

    const int col = n0 + l;
    const float bv = bias ? bias[col] : 0.0f;
    if (accumulate) {
#pragma unroll
        for (int t = 0; t < MT; ++t)
#pragma unroll
            for (int j = 0; j < 8; ++j) {
                const size_t idx = (size_t)(t * 16 + j + 8 * half) * (size_t)ldc + (size_t)col;
                C[idx] += acc[t][j] + bv;
            }
    } else {
#pragma unroll
        for (int t = 0; t < MT; ++t)
#pragma unroll
            for (int j = 0; j < 8; ++j) {
                const size_t idx = (size_t)(t * 16 + j + 8 * half) * (size_t)ldc + (size_t)col;
                C[idx] = acc[t][j] + bv;
            }
    }
}

// ---------------------------------------------------------------------------
// Fused attention score kernel: never materializes enc_t (16.7 MB saved).
// For 16 rows (b,s) per wave: scores[row] = sum_a tanh(enc[row]·W_enc[a]
//     + b_enc[a] + dec_t[b,a]) * W_v[a] + b_v, masked to -1e10.
// encoder_outputs is the only HBM stream; W_enc (2 MB) lives in L2.
// ---------------------------------------------------------------------------
__global__ void attn_score_wmma(const float* __restrict__ enc,     // (B*S, E)
                                const float* __restrict__ W_enc,   // (A, E)
                                const float* __restrict__ b_enc,   // (A)
                                const float* __restrict__ dec_t,   // (B, A)
                                const float* __restrict__ W_v,     // (A)
                                const float* __restrict__ b_v,     // (1)
                                const int*   __restrict__ mask,    // (B*S)
                                float* __restrict__ scores)        // (B*S)
{
    const int wave = (blockIdx.x * blockDim.x + threadIdx.x) >> 5;
    const int lane = threadIdx.x & 31;
    const int half = lane >> 4;
    const int l    = lane & 15;
    const int m0   = wave << 4;          // 16 | S, so all 16 rows share batch b
    const int b    = m0 / Sn;

    const float* arow = enc + (size_t)(m0 + l) * (size_t)En;

    float sc[8] = {0.f, 0.f, 0.f, 0.f, 0.f, 0.f, 0.f, 0.f};

#pragma unroll 1
    for (int n0 = 0; n0 < An; n0 += 16) {
        const float* wrow = W_enc + (size_t)(n0 + l) * (size_t)En;
        v8f acc = {};
#pragma unroll 1
        for (int k = 0; k < En; k += 32) {
            v16bf a, bm;
            load_a16(arow, k, half, a);
            load_b16(wrow, k, half, bm);
            acc = __builtin_amdgcn_wmma_f32_16x16x32_bf16(
                      false, a, false, bm, (short)0, acc, false, false);
        }
        const int col = n0 + l;
        const float d  = dec_t[b * An + col] + b_enc[col];
        const float wv = W_v[col];
#pragma unroll
        for (int j = 0; j < 8; ++j)
            sc[j] += tanh_fast(acc[j] + d) * wv;
    }

    // Reduce over the 16 lanes of each half (cols); lanes 0/16 own the rows.
#pragma unroll
    for (int j = 0; j < 8; ++j) {
        float v = sc[j];
        v += __shfl_xor(v, 1, 32);
        v += __shfl_xor(v, 2, 32);
        v += __shfl_xor(v, 4, 32);
        v += __shfl_xor(v, 8, 32);
        if (l == 0) {
            const int row = m0 + j + 8 * half;
            float s = v + b_v[0];
            if (mask[row] == 0) s = -1e10f;
            scores[row] = s;
        }
    }
}

// ---------------------------------------------------------------------------
// Per-batch softmax over S=128 scores (one block per batch row).
// ---------------------------------------------------------------------------
__global__ void softmax_kernel(const float* __restrict__ scores,
                               float* __restrict__ attn)
{
    __shared__ float red[Sn];
    const int b = blockIdx.x;
    const int t = threadIdx.x;

    const float v = scores[b * Sn + t];
    red[t] = v;
    __syncthreads();
    for (int off = Sn >> 1; off > 0; off >>= 1) {
        if (t < off) red[t] = fmaxf(red[t], red[t + off]);
        __syncthreads();
    }
    const float mx = red[0];
    __syncthreads();

    const float e = __expf(v - mx);
    red[t] = e;
    __syncthreads();
    for (int off = Sn >> 1; off > 0; off >>= 1) {
        if (t < off) red[t] += red[t + off];
        __syncthreads();
    }
    attn[b * Sn + t] = e * rcp_fast(red[0]);
}

// context[b,e] = sum_s attn[b,s] * enc[b,s,e]; grid (B, E/256)
__global__ void context_kernel(const float* __restrict__ attn,
                               const float* __restrict__ enc,
                               float* __restrict__ ctx)
{
    __shared__ float a[Sn];
    const int b = blockIdx.x;
    const int e = blockIdx.y * blockDim.x + threadIdx.x;
    if (threadIdx.x < Sn) a[threadIdx.x] = attn[b * Sn + threadIdx.x];
    __syncthreads();

    const float* ep = enc + (size_t)b * Sn * En + e;
    float acc = 0.f;
#pragma unroll 4
    for (int s = 0; s < Sn; ++s)
        acc += a[s] * ep[(size_t)s * En];
    ctx[b * En + e] = acc;
}

// embedded[b, :] = embedding[input[b], :]
__global__ void embed_gather(const int* __restrict__ input,
                             const float* __restrict__ embedding,
                             float* __restrict__ embedded)
{
    const int idx = blockIdx.x * blockDim.x + threadIdx.x;   // B*EMB threads
    const int b = idx / EMBn;
    const int e = idx - b * EMBn;
    embedded[idx] = embedding[(size_t)input[b] * EMBn + e];
}

// x = [embedded(512), context(1024)] -> (B, 1536)
__global__ void concat_x(const float* __restrict__ embedded,
                         const float* __restrict__ context,
                         float* __restrict__ x)
{
    const int idx = blockIdx.x * blockDim.x + threadIdx.x;   // B*1536 threads
    const int b = idx / (EMBn + En);
    const int j = idx - b * (EMBn + En);
    x[idx] = (j < EMBn) ? embedded[b * EMBn + j] : context[b * En + (j - EMBn)];
}

// pred_in = [h1(1024), context(1024), embedded(512)] -> (B, 2560)
__global__ void concat_pred(const float* __restrict__ h1,
                            const float* __restrict__ context,
                            const float* __restrict__ embedded,
                            float* __restrict__ pred_in)
{
    const int PK = Hn + En + EMBn;
    const int idx = blockIdx.x * blockDim.x + threadIdx.x;   // B*2560 threads
    const int b = idx / PK;
    const int j = idx - b * PK;
    float v;
    if (j < Hn)            v = h1[b * Hn + j];
    else if (j < Hn + En)  v = context[b * En + (j - Hn)];
    else                   v = embedded[b * EMBn + (j - Hn - En)];
    pred_in[idx] = v;
}

// LSTM elementwise: gates (B,4H) layout [i|f|g|o]
__global__ void lstm_elem(const float* __restrict__ gates,
                          const float* __restrict__ c_prev,
                          float* __restrict__ h_out,
                          float* __restrict__ c_out)
{
    const int idx = blockIdx.x * blockDim.x + threadIdx.x;   // B*H threads
    const int b = idx / Hn;
    const int h = idx - b * Hn;
    const float* g = gates + (size_t)b * 4 * Hn;
    const float gi = sigmoid_fast(g[h]);
    const float gf = sigmoid_fast(g[Hn + h]);
    const float gg = tanh_fast(g[2 * Hn + h]);
    const float go = sigmoid_fast(g[3 * Hn + h]);
    const float c  = gf * c_prev[idx] + gi * gg;
    c_out[idx] = c;
    h_out[idx] = go * tanh_fast(c);
}

// ---------------------------------------------------------------------------
extern "C" void kernel_launch(void* const* d_in, const int* in_sizes, int n_in,
                              void* d_out, int out_size, void* d_ws, size_t ws_size,
                              hipStream_t stream)
{
    (void)in_sizes; (void)n_in; (void)out_size; (void)ws_size;

    const int*   input     = (const int*)  d_in[0];
    const float* enc       = (const float*)d_in[1];
    const float* hidden    = (const float*)d_in[2];   // (2, B, H)
    const float* cell      = (const float*)d_in[3];   // (2, B, H)
    const int*   mask      = (const int*)  d_in[4];
    const float* embedding = (const float*)d_in[5];
    const float* W_enc     = (const float*)d_in[6];
    const float* b_enc     = (const float*)d_in[7];
    const float* W_dec     = (const float*)d_in[8];
    const float* b_dec     = (const float*)d_in[9];
    const float* W_v       = (const float*)d_in[10];
    const float* b_v       = (const float*)d_in[11];
    const float* W_ih0     = (const float*)d_in[12];
    const float* W_hh0     = (const float*)d_in[13];
    const float* b_ih0     = (const float*)d_in[14];
    const float* b_hh0     = (const float*)d_in[15];
    const float* W_ih1     = (const float*)d_in[16];
    const float* W_hh1     = (const float*)d_in[17];
    const float* b_ih1     = (const float*)d_in[18];
    const float* b_hh1     = (const float*)d_in[19];
    const float* W_out     = (const float*)d_in[20];
    const float* b_out     = (const float*)d_in[21];

    float* out = (float*)d_out;
    float* ws  = (float*)d_ws;

    // Workspace layout (floats, 16B aligned)
    float* dec_t    = ws;                  //  64*512   = 32768
    float* scores   = ws + 32768;          //  64*128   =  8192
    float* embedded = ws + 40960;          //  64*512   = 32768
    float* context  = ws + 73728;          //  64*1024  = 65536
    float* xbuf     = ws + 139264;         //  64*1536  = 98304
    float* gates0   = ws + 237568;         //  64*4096  = 262144
    float* gates1   = ws + 499712;         //  64*4096  = 262144
    float* pred_in  = ws + 761856;         //  64*2560  = 163840 -> 925696 total

    // Output layout: prediction | new_hidden(2,B,H) | new_cell(2,B,H) | attn
    float* pred = out;                     // 64*32000 = 2048000
    float* h0o  = out + 2048000;
    float* h1o  = h0o + Bn * Hn;
    float* c0o  = out + 2048000 + 2 * Bn * Hn;
    float* c1o  = c0o + Bn * Hn;
    float* attn = out + 2048000 + 4 * Bn * Hn;

    const float* h_prev0 = hidden;                 // hidden[0]
    const float* h_prev1 = hidden + Bn * Hn;       // hidden[1] (top)
    const float* c_prev0 = cell;
    const float* c_prev1 = cell + Bn * Hn;

    // 1) embedding gather
    embed_gather<<<(Bn * EMBn) / 256, 256, 0, stream>>>(input, embedding, embedded);

    // 2) dec_t = top @ W_dec^T + b_dec     (M=64, N=512, K=1024): 32 waves
    gemm_bias_wmma<4><<<4, 256, 0, stream>>>(h_prev1, Hn, W_dec, Hn, b_dec,
                                             dec_t, An, An, Hn, 0);

    // 3) fused attention scores (8192 rows / 16 per wave = 512 waves)
    attn_score_wmma<<<64, 256, 0, stream>>>(enc, W_enc, b_enc, dec_t,
                                            W_v, b_v, mask, scores);

    // 4) softmax -> attention (final output, also consumed by context kernel)
    softmax_kernel<<<Bn, Sn, 0, stream>>>(scores, attn);

    // 5) context = attn @ encoder_outputs
    context_kernel<<<dim3(Bn, En / 256), 256, 0, stream>>>(attn, enc, context);

    // 6) x = [embedded, context]
    concat_x<<<(Bn * (EMBn + En)) / 256, 256, 0, stream>>>(embedded, context, xbuf);

    // 7) gates0 = x @ W_ih0^T + b_ih0 ; += h0 @ W_hh0^T + b_hh0   (N=4096: 256 waves)
    gemm_bias_wmma<4><<<32, 256, 0, stream>>>(xbuf, EMBn + En, W_ih0, EMBn + En, b_ih0,
                                              gates0, 4 * Hn, 4 * Hn, EMBn + En, 0);
    gemm_bias_wmma<4><<<32, 256, 0, stream>>>(h_prev0, Hn, W_hh0, Hn, b_hh0,
                                              gates0, 4 * Hn, 4 * Hn, Hn, 1);

    // 8) LSTM layer 0 elementwise
    lstm_elem<<<(Bn * Hn) / 256, 256, 0, stream>>>(gates0, c_prev0, h0o, c0o);

    // 9) gates1 = h0_new @ W_ih1^T + b_ih1 ; += h1_old @ W_hh1^T + b_hh1
    gemm_bias_wmma<4><<<32, 256, 0, stream>>>(h0o, Hn, W_ih1, Hn, b_ih1,
                                              gates1, 4 * Hn, 4 * Hn, Hn, 0);
    gemm_bias_wmma<4><<<32, 256, 0, stream>>>(h_prev1, Hn, W_hh1, Hn, b_hh1,
                                              gates1, 4 * Hn, 4 * Hn, Hn, 1);

    // 10) LSTM layer 1 elementwise
    lstm_elem<<<(Bn * Hn) / 256, 256, 0, stream>>>(gates1, c_prev1, h1o, c1o);

    // 11) pred_in = [h1, context, embedded]
    concat_pred<<<(Bn * (Hn + En + EMBn)) / 256, 256, 0, stream>>>(h1o, context,
                                                                   embedded, pred_in);

    // 12) prediction = pred_in @ W_out^T + b_out   (M=64, N=32000, K=2560)
    //     2000 waves / 250 blocks; W_out (327 MB) streamed exactly ONCE,
    //     4 WMMAs per B-fragment -> HBM-bound at the ~20 us roofline floor.
    gemm_bias_wmma<4><<<250, 256, 0, stream>>>(pred_in, Hn + En + EMBn,
                                               W_out, Hn + En + EMBn, b_out,
                                               pred, Vn, Vn, Hn + En + EMBn, 0);
}
